// MMAEKNN_42563125903677
// MI455X (gfx1250) — compile-verified, matrix operand order checked by reading, not run on hardware
//
#include <hip/hip_runtime.h>
#include <hip/hip_bf16.h>
#include <stdint.h>

// ---------------------------------------------------------------------------
// MMAEKNN for MI455X (gfx1250): bf16 WMMA GEMM pipeline + KNN loss
//  - all GEMMs / gram matrices via v_wmma_f32_16x16x32_bf16 (wave32)
//  - tile staging global->LDS via Tensor Data Mover, double-buffered:
//    wave0 issues DMA for tile k+1, waits TENSORcnt<=2 (in-order completion
//    guarantees tile k landed), WMMA overlaps the in-flight DMA.
// ---------------------------------------------------------------------------

typedef __attribute__((ext_vector_type(16))) __bf16          v16bf;
typedef __attribute__((ext_vector_type(8)))  float           v8f;
typedef __attribute__((ext_vector_type(4)))  unsigned int    v4u;
typedef __attribute__((ext_vector_type(8)))  unsigned int    v8u;

#define BB   8192     // batch rows
#define DD   1024     // input dim
#define HH1  2048
#define HH2  512
#define LL   64       // latent dim
#define KNN  15

#define ZN        (BB * LL)
#define OUT_TOTAL 0
#define OUT_Z     1
#define OUT_REC   (1 + ZN)
#define OUT_KNN   (2 + ZN)

#define TM   128
#define TN   128
#define TK   32
#define ASTR 40      // bf16 elems/row in LDS: 64B data + 16B pad (16B aligned)

#if defined(__gfx1250__)
#define USE_TDM 1
#else
#define USE_TDM 0
#endif

#if defined(__gfx1250__) && __has_builtin(__builtin_amdgcn_s_wait_tensorcnt)
#define TDM_WAIT(n) __builtin_amdgcn_s_wait_tensorcnt(n)
#elif defined(__gfx1250__)
#define TDM_WAIT(n) asm volatile("s_wait_tensorcnt %0" :: "i"(n) : "memory")
#else
#define TDM_WAIT(n)
#endif

__device__ __forceinline__ unsigned short f32_to_bf16_bits(float f) {
  unsigned int u = __float_as_uint(f);
  unsigned int r = u + 0x7FFFu + ((u >> 16) & 1u);   // round-to-nearest-even
  return (unsigned short)(r >> 16);
}

#if USE_TDM
// ---------------------------------------------------------------------------
// TDM: DMA a (rows x TK) bf16 tile (row stride = row_stride_elems) into LDS
// laid out with ASTR padding. D# per CDNA5 ISA ch.8:
//   g0: count=1 | lds_addr | global_addr(57b) | type=2
//   g1: data_size=2B, pad_enable, pad_interval=16DW, pad_amount=4DW,
//       tensor_dim0/1 (OOB rows/cols read as zero), tile 32 x 128, stride.
// Issued wave-uniform (2D tensor: descriptor groups 0 and 1 only).
// ---------------------------------------------------------------------------
__device__ __forceinline__ void tdm_load_rows_tile(
    unsigned lds_byte_off, const void* gptr, unsigned rows_avail,
    unsigned k_avail, unsigned row_stride_elems) {
  const unsigned long long ga = (unsigned long long)(uintptr_t)gptr;
  v4u g0;
  g0[0] = 1u;                                            // count = 1 (valid)
  g0[1] = lds_byte_off;                                  // lds_addr
  g0[2] = (unsigned)(ga & 0xffffffffu);                  // global_addr lo
  g0[3] = (unsigned)((ga >> 32) & 0x01ffffffu) | (2u << 30);  // hi | type=2
  v8u g1;
  g1[0] = (1u << 16)      // data_size = 1 -> 2 bytes
        | (1u << 20)      // pad_enable
        | (3u << 22)      // pad_interval = 3 -> every 16 DWORDs (64B row)
        | (3u << 25);     // pad_amount   = 3 -> 4 DWORDs (16B pad)
  g1[1] = (k_avail & 0xffffu) << 16;                     // tensor_dim0[15:0]
  g1[2] = ((k_avail >> 16) & 0xffffu)                    // tensor_dim0[31:16]
        | ((rows_avail & 0xffffu) << 16);                // tensor_dim1[15:0]
  g1[3] = ((rows_avail >> 16) & 0xffffu)                 // tensor_dim1[31:16]
        | ((unsigned)TK << 16);                          // tile_dim0 = 32
  g1[4] = (unsigned)TM;                                  // tile_dim1 = 128
  g1[5] = row_stride_elems;                              // dim0_stride[31:0]
  g1[6] = 0u;                                            // stride hi bits
  g1[7] = 0u;
  asm volatile("tensor_load_to_lds %0, %1" :: "s"(g0), "s"(g1) : "memory");
}
#endif

// ---------------------------------------------------------------------------
// f32 -> bf16 cast, grid-stride
// ---------------------------------------------------------------------------
__global__ void cast_bf16_kernel(const float* __restrict__ src,
                                 unsigned short* __restrict__ dst, int n) {
  for (int i = blockIdx.x * blockDim.x + threadIdx.x; i < n;
       i += gridDim.x * blockDim.x)
    dst[i] = f32_to_bf16_bits(src[i]);
}

// ---------------------------------------------------------------------------
// f32 [K x N] -> bf16 [N x K] transpose-cast (weights), LDS 32x32 tiles
// ---------------------------------------------------------------------------
__global__ __launch_bounds__(256) void cast_transpose_bf16_kernel(
    const float* __restrict__ src, unsigned short* __restrict__ dst,
    int K, int N) {
  __shared__ float tile[32][33];
  const int kb = blockIdx.x * 32, nb = blockIdx.y * 32;
  const int tx = threadIdx.x & 31, ty = threadIdx.x >> 5;   // 32 x 8
#pragma unroll
  for (int r = ty; r < 32; r += 8) {
    const int k = kb + r, n = nb + tx;
    tile[r][tx] = (k < K && n < N) ? src[(size_t)k * N + n] : 0.f;
  }
  __syncthreads();
#pragma unroll
  for (int r = ty; r < 32; r += 8) {
    const int n = nb + r, k = kb + tx;
    if (n < N && k < K) dst[(size_t)n * K + k] = f32_to_bf16_bits(tile[tx][r]);
  }
}

// ---------------------------------------------------------------------------
// WMMA fragment load from a rows x TK LDS tile (stride ASTR), per ISA §7.12.2:
// 16-bit operand 16x32: lanes 0-15 hold K {0..7,16..23}, lanes 16-31 +8.
// Both 8-halfword runs are contiguous -> two b128 LDS loads.
// ---------------------------------------------------------------------------
__device__ __forceinline__ v16bf load_frag_rowk(const unsigned short* S,
                                                int row, int lane) {
  const int kb = (lane >> 4) << 3;   // 0 or 8
  union { uint4 q[2]; v16bf b; } cv;
  cv.q[0] = *(const uint4*)&S[row * ASTR + kb];        // K = kb .. kb+7
  cv.q[1] = *(const uint4*)&S[row * ASTR + kb + 16];   // K = kb+16 .. kb+23
  return cv.b;
}

// ---------------------------------------------------------------------------
// GEMM: C = act(A[MxK]_bf16 @ Wt[NxK]_bf16^T + bias), 128x128 tile, 8 waves,
// double-buffered TDM staging.
// ---------------------------------------------------------------------------
__global__ __launch_bounds__(256) void gemm_bias_act_wmma(
    const unsigned short* __restrict__ A, const unsigned short* __restrict__ Wt,
    const float* __restrict__ bias, float* __restrict__ Cf,
    unsigned short* __restrict__ Cb, int M, int N, int K, int relu) {
  __shared__ unsigned short As[2][TM * ASTR];
  __shared__ unsigned short Bs[2][TN * ASTR];
  const int tid  = threadIdx.x;
  const int lane = tid & 31;
  const int wave = tid >> 5;
  const int wrow = wave & 3;    // 4 waves down  (32 rows each)
  const int wcol = wave >> 2;   // 2 waves across (64 cols each)
  const int m0 = blockIdx.y * TM;
  const int n0 = blockIdx.x * TN;
  const int nk = K / TK;

  const v8f zinit = {};
  v8f acc[2][4];
#pragma unroll
  for (int i = 0; i < 2; ++i)
#pragma unroll
    for (int j = 0; j < 4; ++j) acc[i][j] = zinit;

  const int lr = tid >> 1, lc = (tid & 1) << 4;  // 2 threads/row, 16 elems

#if USE_TDM
  if (wave == 0) {   // prologue: DMA tile 0 into buffer 0
    tdm_load_rows_tile((unsigned)(uintptr_t)&As[0][0], A + (size_t)m0 * K,
                       (unsigned)(M - m0), (unsigned)K, (unsigned)K);
    tdm_load_rows_tile((unsigned)(uintptr_t)&Bs[0][0], Wt + (size_t)n0 * K,
                       (unsigned)(N - n0), (unsigned)K, (unsigned)K);
  }
#endif

  for (int it = 0; it < nk; ++it) {
    const int cur = it & 1;
    const int k0 = it * TK;
#if USE_TDM
    if (wave == 0) {
      if (it + 1 < nk) {   // DMA next tile into other buffer, keep 2 in flight
        const int kn = k0 + TK;
        tdm_load_rows_tile((unsigned)(uintptr_t)&As[cur ^ 1][0],
                           A + (size_t)m0 * K + kn,
                           (unsigned)(M - m0), (unsigned)(K - kn), (unsigned)K);
        tdm_load_rows_tile((unsigned)(uintptr_t)&Bs[cur ^ 1][0],
                           Wt + (size_t)n0 * K + kn,
                           (unsigned)(N - n0), (unsigned)(K - kn), (unsigned)K);
        TDM_WAIT(2);   // in-order: oldest pair (tile it) has landed
      } else {
        TDM_WAIT(0);
      }
    }
#else
    {
      const unsigned short* src = A + (size_t)(m0 + lr) * K + (k0 + lc);
      *(uint4*)&As[cur][lr * ASTR + lc]     = *(const uint4*)(src);
      *(uint4*)&As[cur][lr * ASTR + lc + 8] = *(const uint4*)(src + 8);
    }
    if (n0 + lr < N) {
      const unsigned short* src = Wt + (size_t)(n0 + lr) * K + (k0 + lc);
      *(uint4*)&Bs[cur][lr * ASTR + lc]     = *(const uint4*)(src);
      *(uint4*)&Bs[cur][lr * ASTR + lc + 8] = *(const uint4*)(src + 8);
    } else {
      const uint4 zz = {0u, 0u, 0u, 0u};
      *(uint4*)&Bs[cur][lr * ASTR + lc]     = zz;
      *(uint4*)&Bs[cur][lr * ASTR + lc + 8] = zz;
    }
#endif
    __syncthreads();

    v16bf afrag[2], bfrag[4];
#pragma unroll
    for (int fr = 0; fr < 2; ++fr)
      afrag[fr] =
          load_frag_rowk(As[cur], wrow * 32 + fr * 16 + (lane & 15), lane);
#pragma unroll
    for (int fc = 0; fc < 4; ++fc)
      bfrag[fc] =
          load_frag_rowk(Bs[cur], wcol * 64 + fc * 16 + (lane & 15), lane);

#pragma unroll
    for (int fr = 0; fr < 2; ++fr)
#pragma unroll
      for (int fc = 0; fc < 4; ++fc)
        acc[fr][fc] = __builtin_amdgcn_wmma_f32_16x16x32_bf16(
            false, afrag[fr], false, bfrag[fc], (short)0, acc[fr][fc],
            false, false);
    __syncthreads();   // all waves done reading buf[cur] before it is reused
  }

#pragma unroll
  for (int fr = 0; fr < 2; ++fr) {
#pragma unroll
    for (int fc = 0; fc < 4; ++fc) {
      const int colg = n0 + wcol * 64 + fc * 16 + (lane & 15);
#pragma unroll
      for (int i = 0; i < 8; ++i) {
        const int rowg = m0 + wrow * 32 + fr * 16 + ((lane >> 4) << 3) + i;
        if (rowg < M && colg < N) {
          float v = acc[fr][fc][i] + bias[colg];
          if (relu) v = fmaxf(v, 0.f);
          if (Cf) Cf[(size_t)rowg * N + colg] = v;
          if (Cb) Cb[(size_t)rowg * N + colg] = f32_to_bf16_bits(v);
        }
      }
    }
  }
}

// ---------------------------------------------------------------------------
// Gram->distance: Dist[i][j] = sqrt(max(0, sq[i]+sq[j]-2*A_i.A_j))
// A@A^T: both operand tiles are row-tiles of A. Double-buffered TDM staging.
// ---------------------------------------------------------------------------
__global__ __launch_bounds__(256) void gram_dist_wmma(
    const unsigned short* __restrict__ Ab, const float* __restrict__ sq,
    float* __restrict__ Dist, int Bn, int K) {
  __shared__ unsigned short As[2][TM * ASTR];
  __shared__ unsigned short Bs[2][TN * ASTR];
  const int tid  = threadIdx.x;
  const int lane = tid & 31;
  const int wave = tid >> 5;
  const int wrow = wave & 3;
  const int wcol = wave >> 2;
  const int m0 = blockIdx.y * TM;
  const int n0 = blockIdx.x * TN;
  const int nk = K / TK;

  const v8f zinit = {};
  v8f acc[2][4];
#pragma unroll
  for (int i = 0; i < 2; ++i)
#pragma unroll
    for (int j = 0; j < 4; ++j) acc[i][j] = zinit;

  const int lr = tid >> 1, lc = (tid & 1) << 4;

#if USE_TDM
  if (wave == 0) {
    tdm_load_rows_tile((unsigned)(uintptr_t)&As[0][0], Ab + (size_t)m0 * K,
                       (unsigned)(Bn - m0), (unsigned)K, (unsigned)K);
    tdm_load_rows_tile((unsigned)(uintptr_t)&Bs[0][0], Ab + (size_t)n0 * K,
                       (unsigned)(Bn - n0), (unsigned)K, (unsigned)K);
  }
#endif

  for (int it = 0; it < nk; ++it) {
    const int cur = it & 1;
    const int k0 = it * TK;
#if USE_TDM
    if (wave == 0) {
      if (it + 1 < nk) {
        const int kn = k0 + TK;
        tdm_load_rows_tile((unsigned)(uintptr_t)&As[cur ^ 1][0],
                           Ab + (size_t)m0 * K + kn,
                           (unsigned)(Bn - m0), (unsigned)(K - kn), (unsigned)K);
        tdm_load_rows_tile((unsigned)(uintptr_t)&Bs[cur ^ 1][0],
                           Ab + (size_t)n0 * K + kn,
                           (unsigned)(Bn - n0), (unsigned)(K - kn), (unsigned)K);
        TDM_WAIT(2);
      } else {
        TDM_WAIT(0);
      }
    }
#else
    {
      const unsigned short* src = Ab + (size_t)(m0 + lr) * K + (k0 + lc);
      *(uint4*)&As[cur][lr * ASTR + lc]     = *(const uint4*)(src);
      *(uint4*)&As[cur][lr * ASTR + lc + 8] = *(const uint4*)(src + 8);
    }
    {
      const unsigned short* src = Ab + (size_t)(n0 + lr) * K + (k0 + lc);
      *(uint4*)&Bs[cur][lr * ASTR + lc]     = *(const uint4*)(src);
      *(uint4*)&Bs[cur][lr * ASTR + lc + 8] = *(const uint4*)(src + 8);
    }
#endif
    __syncthreads();

    v16bf afrag[2], bfrag[4];
#pragma unroll
    for (int fr = 0; fr < 2; ++fr)
      afrag[fr] =
          load_frag_rowk(As[cur], wrow * 32 + fr * 16 + (lane & 15), lane);
#pragma unroll
    for (int fc = 0; fc < 4; ++fc)
      bfrag[fc] =
          load_frag_rowk(Bs[cur], wcol * 64 + fc * 16 + (lane & 15), lane);

#pragma unroll
    for (int fr = 0; fr < 2; ++fr)
#pragma unroll
      for (int fc = 0; fc < 4; ++fc)
        acc[fr][fc] = __builtin_amdgcn_wmma_f32_16x16x32_bf16(
            false, afrag[fr], false, bfrag[fc], (short)0, acc[fr][fc],
            false, false);
    __syncthreads();
  }

#pragma unroll
  for (int fr = 0; fr < 2; ++fr) {
#pragma unroll
    for (int fc = 0; fc < 4; ++fc) {
      const int colg = n0 + wcol * 64 + fc * 16 + (lane & 15);
#pragma unroll
      for (int i = 0; i < 8; ++i) {
        const int rowg = m0 + wrow * 32 + fr * 16 + ((lane >> 4) << 3) + i;
        const float d2 = sq[rowg] + sq[colg] - 2.f * acc[fr][fc][i];
        const float d  = (d2 > 0.f) ? sqrtf(d2) : 0.f;
        Dist[(size_t)rowg * Bn + colg] = d;
      }
    }
  }
}

// ---------------------------------------------------------------------------
// Row squared norms (f32 source for accuracy)
// ---------------------------------------------------------------------------
__global__ __launch_bounds__(256) void row_sqnorm_kernel(
    const float* __restrict__ A, int K, float* __restrict__ sq) {
  __shared__ float red[256];
  const int row = blockIdx.x;
  float s = 0.f;
  for (int k = threadIdx.x; k < K; k += 256) {
    const float v = A[(size_t)row * K + k];
    s += v * v;
  }
  red[threadIdx.x] = s;
  __syncthreads();
  for (int off = 128; off > 0; off >>= 1) {
    if (threadIdx.x < off) red[threadIdx.x] += red[threadIdx.x + off];
    __syncthreads();
  }
  if (threadIdx.x == 0) sq[row] = red[0];
}

// ---------------------------------------------------------------------------
// Top-15 smallest per row (diag excluded). One wave32 per row; per-lane
// sorted lists merged by 15 rounds of 64-bit keyed argmin over __shfl_xor.
// ---------------------------------------------------------------------------
__global__ __launch_bounds__(256) void topk_kernel(
    const float* __restrict__ Dist, int Bn, int* __restrict__ knn_idx) {
  const int lane = threadIdx.x & 31;
  const int row  = blockIdx.x * 8 + (threadIdx.x >> 5);
  if (row >= Bn) return;
  const float* dr = Dist + (size_t)row * Bn;

  float vals[KNN];
  int   idxs[KNN];
#pragma unroll
  for (int s = 0; s < KNN; ++s) {
    vals[s] = __int_as_float(0x7f800000);
    idxs[s] = 0x7fffffff;
  }

  for (int j = lane; j < Bn; j += 32) {
    const float d = (j == row) ? __int_as_float(0x7f800000) : dr[j];
    if (d < vals[KNN - 1]) {
      int p = KNN - 1;
      while (p > 0 && vals[p - 1] > d) {
        vals[p] = vals[p - 1]; idxs[p] = idxs[p - 1]; --p;
      }
      vals[p] = d; idxs[p] = j;
    }
  }

  int ptr = 0;
  for (int s = 0; s < KNN; ++s) {
    const float myv = (ptr < KNN) ? vals[ptr] : __int_as_float(0x7f800000);
    const unsigned int myj = (ptr < KNN) ? (unsigned int)idxs[ptr] : 0x7fffffffu;
    unsigned long long key =
        ((unsigned long long)__float_as_uint(myv) << 32) | myj;
    for (int off = 16; off > 0; off >>= 1) {
      const unsigned long long o = __shfl_xor(key, off, 32);
      if (o < key) key = o;
    }
    const int jwin = (int)(key & 0xffffffffu);
    if (ptr < KNN && idxs[ptr] == jwin) ++ptr;
    if (lane == 0) knn_idx[row * KNN + s] = jwin;
  }
}

// ---------------------------------------------------------------------------
// Mask bit-matrix (8192x8192 bits = 8 MB): zero, then symmetric scatter.
// ---------------------------------------------------------------------------
__global__ void zero_u32_kernel(unsigned int* __restrict__ p, size_t n) {
  for (size_t i = blockIdx.x * (size_t)blockDim.x + threadIdx.x; i < n;
       i += (size_t)gridDim.x * blockDim.x)
    p[i] = 0u;
}

__global__ void scatter_mask_kernel(const int* __restrict__ knn,
                                    unsigned int* __restrict__ mask) {
  const int t = blockIdx.x * blockDim.x + threadIdx.x;
  if (t >= BB * KNN) return;
  const int i = t / KNN;
  const int j = knn[t];
  atomicOr(&mask[(size_t)i * (BB / 32) + (j >> 5)], 1u << (j & 31));
  atomicOr(&mask[(size_t)j * (BB / 32) + (i >> 5)], 1u << (i & 31));
}

// ---------------------------------------------------------------------------
// Masked max of both distance matrices -> per-block partials (deterministic)
// ---------------------------------------------------------------------------
__global__ __launch_bounds__(256) void masked_max_kernel(
    const unsigned int* __restrict__ mask, const float* __restrict__ XD,
    const float* __restrict__ ZD, float* __restrict__ pmx,
    float* __restrict__ pmz) {
  const size_t nwords = (size_t)BB * (BB / 32);
  float mx = 0.f, mz = 0.f;
  for (size_t w = blockIdx.x * (size_t)blockDim.x + threadIdx.x; w < nwords;
       w += (size_t)gridDim.x * blockDim.x) {
    unsigned int bits = mask[w];
    while (bits) {
      const int b = __ffs(bits) - 1;
      bits &= bits - 1;
      const size_t idx = w * 32 + b;
      mx = fmaxf(mx, XD[idx]);
      mz = fmaxf(mz, ZD[idx]);
    }
  }
  __shared__ float rx[256], rz[256];
  rx[threadIdx.x] = mx; rz[threadIdx.x] = mz;
  __syncthreads();
  for (int off = 128; off > 0; off >>= 1) {
    if (threadIdx.x < off) {
      rx[threadIdx.x] = fmaxf(rx[threadIdx.x], rx[threadIdx.x + off]);
      rz[threadIdx.x] = fmaxf(rz[threadIdx.x], rz[threadIdx.x + off]);
    }
    __syncthreads();
  }
  if (threadIdx.x == 0) { pmx[blockIdx.x] = rx[0]; pmz[blockIdx.x] = rz[0]; }
}

__global__ __launch_bounds__(256) void reduce_max_kernel(
    const float* __restrict__ pmx, const float* __restrict__ pmz, int n,
    float* __restrict__ smax) {
  __shared__ float rx[256], rz[256];
  float mx = 0.f, mz = 0.f;
  for (int i = threadIdx.x; i < n; i += 256) {
    mx = fmaxf(mx, pmx[i]); mz = fmaxf(mz, pmz[i]);
  }
  rx[threadIdx.x] = mx; rz[threadIdx.x] = mz;
  __syncthreads();
  for (int off = 128; off > 0; off >>= 1) {
    if (threadIdx.x < off) {
      rx[threadIdx.x] = fmaxf(rx[threadIdx.x], rx[threadIdx.x + off]);
      rz[threadIdx.x] = fmaxf(rz[threadIdx.x], rz[threadIdx.x + off]);
    }
    __syncthreads();
  }
  if (threadIdx.x == 0) { smax[0] = rx[0]; smax[1] = rz[0]; }
}

// ---------------------------------------------------------------------------
// Masked sum of (z_norm - x_norm)^2 + mask count -> partials
// ---------------------------------------------------------------------------
__global__ __launch_bounds__(256) void masked_sum_kernel(
    const unsigned int* __restrict__ mask, const float* __restrict__ XD,
    const float* __restrict__ ZD, const float* __restrict__ smax,
    float* __restrict__ ps, unsigned int* __restrict__ pcnt) {
  const float xinv = 1.f / (smax[0] + 1e-8f);
  const float zinv = 1.f / (smax[1] + 1e-8f);
  const size_t nwords = (size_t)BB * (BB / 32);
  float s = 0.f;
  unsigned int c = 0;
  for (size_t w = blockIdx.x * (size_t)blockDim.x + threadIdx.x; w < nwords;
       w += (size_t)gridDim.x * blockDim.x) {
    unsigned int bits = mask[w];
    while (bits) {
      const int b = __ffs(bits) - 1;
      bits &= bits - 1;
      const size_t idx = w * 32 + b;
      const float d = ZD[idx] * zinv - XD[idx] * xinv;
      s += d * d; ++c;
    }
  }
  __shared__ float rs[256];
  __shared__ unsigned int rc[256];
  rs[threadIdx.x] = s; rc[threadIdx.x] = c;
  __syncthreads();
  for (int off = 128; off > 0; off >>= 1) {
    if (threadIdx.x < off) {
      rs[threadIdx.x] += rs[threadIdx.x + off];
      rc[threadIdx.x] += rc[threadIdx.x + off];
    }
    __syncthreads();
  }
  if (threadIdx.x == 0) { ps[blockIdx.x] = rs[0]; pcnt[blockIdx.x] = rc[0]; }
}

// ---------------------------------------------------------------------------
// Reconstruction-loss partials: sum (x_rec - x)^2
// ---------------------------------------------------------------------------
__global__ __launch_bounds__(256) void rec_partial_kernel(
    const float* __restrict__ x, const float* __restrict__ xr,
    float* __restrict__ pr) {
  const size_t n = (size_t)BB * DD;
  float s = 0.f;
  for (size_t i = blockIdx.x * (size_t)blockDim.x + threadIdx.x; i < n;
       i += (size_t)gridDim.x * blockDim.x) {
    const float d = xr[i] - x[i];
    s += d * d;
  }
  __shared__ float rs[256];
  rs[threadIdx.x] = s;
  __syncthreads();
  for (int off = 128; off > 0; off >>= 1) {
    if (threadIdx.x < off) rs[threadIdx.x] += rs[threadIdx.x + off];
    __syncthreads();
  }
  if (threadIdx.x == 0) pr[blockIdx.x] = rs[0];
}

// ---------------------------------------------------------------------------
// Output assembly
// ---------------------------------------------------------------------------
__global__ void copy_z_kernel(const float* __restrict__ z,
                              float* __restrict__ out) {
  for (int i = blockIdx.x * blockDim.x + threadIdx.x; i < ZN;
       i += gridDim.x * blockDim.x)
    out[OUT_Z + i] = z[i];
}

__global__ __launch_bounds__(256) void finalize_kernel(
    const float* __restrict__ ps, const unsigned int* __restrict__ pcnt,
    const float* __restrict__ pr, int n, float* __restrict__ out) {
  __shared__ float rs[256], rr[256];
  __shared__ unsigned int rc[256];
  float s = 0.f, r = 0.f;
  unsigned int c = 0;
  for (int i = threadIdx.x; i < n; i += 256) {
    s += ps[i]; r += pr[i]; c += pcnt[i];
  }
  rs[threadIdx.x] = s; rr[threadIdx.x] = r; rc[threadIdx.x] = c;
  __syncthreads();
  for (int off = 128; off > 0; off >>= 1) {
    if (threadIdx.x < off) {
      rs[threadIdx.x] += rs[threadIdx.x + off];
      rr[threadIdx.x] += rr[threadIdx.x + off];
      rc[threadIdx.x] += rc[threadIdx.x + off];
    }
    __syncthreads();
  }
  if (threadIdx.x == 0) {
    const float rec = rr[0] / ((float)BB * (float)DD);
    const float knn = rc[0] ? rs[0] / (float)rc[0] : 0.f;
    out[OUT_TOTAL] = rec + knn;
    out[OUT_REC]   = rec;
    out[OUT_KNN]   = knn;
  }
}

// ---------------------------------------------------------------------------
// Host launcher
// ---------------------------------------------------------------------------
extern "C" void kernel_launch(void* const* d_in, const int* in_sizes, int n_in,
                              void* d_out, int out_size, void* d_ws,
                              size_t ws_size, hipStream_t stream) {
  const float* x   = (const float*)d_in[0];
  const float* we0 = (const float*)d_in[1];
  const float* be0 = (const float*)d_in[2];
  const float* we1 = (const float*)d_in[3];
  const float* be1 = (const float*)d_in[4];
  const float* we2 = (const float*)d_in[5];
  const float* be2 = (const float*)d_in[6];
  const float* wd0 = (const float*)d_in[7];
  const float* bd0 = (const float*)d_in[8];
  const float* wd1 = (const float*)d_in[9];
  const float* bd1 = (const float*)d_in[10];
  const float* wd2 = (const float*)d_in[11];
  const float* bd2 = (const float*)d_in[12];
  float* out = (float*)d_out;

  char* wsp = (char*)d_ws;
  auto alloc = [&](size_t bytes) -> void* {
    void* p = (void*)wsp;
    wsp += (bytes + 255) & ~(size_t)255;
    return p;
  };

  unsigned short* xb   = (unsigned short*)alloc((size_t)BB * DD * 2);
  unsigned short* we0t = (unsigned short*)alloc((size_t)DD * HH1 * 2);
  unsigned short* we1t = (unsigned short*)alloc((size_t)HH1 * HH2 * 2);
  unsigned short* we2t = (unsigned short*)alloc((size_t)HH2 * LL * 2);
  unsigned short* wd0t = (unsigned short*)alloc((size_t)LL * HH2 * 2);
  unsigned short* wd1t = (unsigned short*)alloc((size_t)HH2 * HH1 * 2);
  unsigned short* wd2t = (unsigned short*)alloc((size_t)HH1 * DD * 2);
  unsigned short* h1b  = (unsigned short*)alloc((size_t)BB * HH1 * 2);
  unsigned short* h2b  = (unsigned short*)alloc((size_t)BB * HH2 * 2);
  float*          zf   = (float*)alloc((size_t)BB * LL * 4);
  unsigned short* zb   = (unsigned short*)alloc((size_t)BB * LL * 2);
  unsigned short* d0b  = (unsigned short*)alloc((size_t)BB * HH2 * 2);
  unsigned short* d1b  = (unsigned short*)alloc((size_t)BB * HH1 * 2);
  float*          xrec = (float*)alloc((size_t)BB * DD * 4);
  float*          sqx  = (float*)alloc((size_t)BB * 4);
  float*          sqz  = (float*)alloc((size_t)BB * 4);
  float*          XD   = (float*)alloc((size_t)BB * BB * 4);
  float*          ZD   = (float*)alloc((size_t)BB * BB * 4);
  int*            knn  = (int*)alloc((size_t)BB * KNN * 4);
  unsigned int*   mask = (unsigned int*)alloc((size_t)BB * (BB / 32) * 4);
  float*          pmx  = (float*)alloc(1024 * 4);
  float*          pmz  = (float*)alloc(1024 * 4);
  float*          smax = (float*)alloc(2 * 4);
  float*          ps   = (float*)alloc(1024 * 4);
  unsigned int*   pcnt = (unsigned int*)alloc(1024 * 4);
  float*          pr   = (float*)alloc(1024 * 4);

  // 1) activation cast + weight transpose-casts (weights: [K x N] -> [N x K])
  cast_bf16_kernel<<<2048, 256, 0, stream>>>(x, xb, BB * DD);
  cast_transpose_bf16_kernel<<<dim3(DD / 32, HH1 / 32), 256, 0, stream>>>(
      we0, we0t, DD, HH1);
  cast_transpose_bf16_kernel<<<dim3(HH1 / 32, HH2 / 32), 256, 0, stream>>>(
      we1, we1t, HH1, HH2);
  cast_transpose_bf16_kernel<<<dim3(HH2 / 32, LL / 32), 256, 0, stream>>>(
      we2, we2t, HH2, LL);
  cast_transpose_bf16_kernel<<<dim3(LL / 32, HH2 / 32), 256, 0, stream>>>(
      wd0, wd0t, LL, HH2);
  cast_transpose_bf16_kernel<<<dim3(HH2 / 32, HH1 / 32), 256, 0, stream>>>(
      wd1, wd1t, HH2, HH1);
  cast_transpose_bf16_kernel<<<dim3(HH1 / 32, DD / 32), 256, 0, stream>>>(
      wd2, wd2t, HH1, DD);

  // 2) MLP forward via bf16 WMMA GEMMs (double-buffered TDM tiles)
  gemm_bias_act_wmma<<<dim3(HH1 / TN, BB / TM), 256, 0, stream>>>(
      xb, we0t, be0, nullptr, h1b, BB, HH1, DD, 1);
  gemm_bias_act_wmma<<<dim3(HH2 / TN, BB / TM), 256, 0, stream>>>(
      h1b, we1t, be1, nullptr, h2b, BB, HH2, HH1, 1);
  gemm_bias_act_wmma<<<dim3(1, BB / TM), 256, 0, stream>>>(
      h2b, we2t, be2, zf, zb, BB, LL, HH2, 0);          // z (no relu)
  gemm_bias_act_wmma<<<dim3(HH2 / TN, BB / TM), 256, 0, stream>>>(
      zb, wd0t, bd0, nullptr, d0b, BB, HH2, LL, 1);
  gemm_bias_act_wmma<<<dim3(HH1 / TN, BB / TM), 256, 0, stream>>>(
      d0b, wd1t, bd1, nullptr, d1b, BB, HH1, HH2, 1);
  gemm_bias_act_wmma<<<dim3(DD / TN, BB / TM), 256, 0, stream>>>(
      d1b, wd2t, bd2, xrec, nullptr, BB, DD, HH1, 0);   // x_rec (no relu)

  // 3) pairwise distances (WMMA grams + norm trick)
  row_sqnorm_kernel<<<BB, 256, 0, stream>>>(x, DD, sqx);
  row_sqnorm_kernel<<<BB, 256, 0, stream>>>(zf, LL, sqz);
  gram_dist_wmma<<<dim3(BB / TN, BB / TM), 256, 0, stream>>>(xb, sqx, XD, BB, DD);
  gram_dist_wmma<<<dim3(BB / TN, BB / TM), 256, 0, stream>>>(zb, sqz, ZD, BB, LL);

  // 4) top-15 per row, symmetric mask
  zero_u32_kernel<<<2048, 256, 0, stream>>>(mask, (size_t)BB * (BB / 32));
  topk_kernel<<<BB / 8, 256, 0, stream>>>(XD, BB, knn);
  scatter_mask_kernel<<<(BB * KNN + 255) / 256, 256, 0, stream>>>(knn, mask);

  // 5) masked maxes, masked mean, rec loss (deterministic 2-stage)
  masked_max_kernel<<<1024, 256, 0, stream>>>(mask, XD, ZD, pmx, pmz);
  reduce_max_kernel<<<1, 256, 0, stream>>>(pmx, pmz, 1024, smax);
  masked_sum_kernel<<<1024, 256, 0, stream>>>(mask, XD, ZD, smax, ps, pcnt);
  rec_partial_kernel<<<1024, 256, 0, stream>>>(x, xrec, pr);

  // 6) outputs: total, z, rec, knn
  copy_z_kernel<<<2048, 256, 0, stream>>>(zf, out);
  finalize_kernel<<<1, 256, 0, stream>>>(ps, pcnt, pr, 1024, out);
}
